// FixupBottleneck_49830210568337
// MI455X (gfx1250) — matrix operand order repaired and checked by reading, not compiled, not run on hardware
//
#include <hip/hip_runtime.h>
#include <stdint.h>

typedef __bf16 bf16_t;
typedef bf16_t v16bf __attribute__((ext_vector_type(16)));
typedef bf16_t v8bf  __attribute__((ext_vector_type(8)));
typedef bf16_t v4bf  __attribute__((ext_vector_type(4)));
typedef float  v8f   __attribute__((ext_vector_type(8)));

#define NPIX 25088   // 128 * 14 * 14
#define HW   196
#define WD   14

// ---------------------------------------------------------------------------
// WMMA wrapper: D = A(16x32 bf16) * B(32x16 bf16) + C(16x16 f32)
// ---------------------------------------------------------------------------
__device__ __forceinline__ v8f wmma_bf16(v16bf a, v16bf b, v8f c) {
    return __builtin_amdgcn_wmma_f32_16x16x32_bf16(
        false, a, false, b, (short)0, c, false, false);
}

// A fragment per ISA 16-bit A layout (16x32 MxK):
//   lanes 0-15 : row M=lane,    elems 0..7 = K 0..7,  elems 8..15 = K 16..23
//   lanes16-31 : row M=lane-16, elems 0..7 = K 8..15, elems 8..15 = K 24..31
__device__ __forceinline__ v16bf load_a_frag(const bf16_t* p, int ka, int kb) {
    v8bf lo = *(const v8bf*)(p + ka);
    v8bf hi = *(const v8bf*)(p + kb);
    return __builtin_shufflevector(lo, hi, 0,1,2,3,4,5,6,7,8,9,10,11,12,13,14,15);
}

// CDNA5 async copy: global -> LDS, tracked by ASYNCcnt (ISA 08_async_tensor §4).
// VDST = LDS byte address (wave-relative = low 32 bits of the generic pointer,
// per the ISA flat->LDS truncation rule), VADDR = 64-bit global address.
__device__ __forceinline__ void async_ld_b128(uint32_t lds_addr, const void* g) {
    asm volatile("global_load_async_to_lds_b128 %0, %1, off"
                 :: "v"(lds_addr), "v"(g) : "memory");
}

// Stage one 8KB B k-slice (256 threads x 2 x 16B).
__device__ __forceinline__ void stage_btile(uint32_t lds_dst, const char* src, int t) {
    async_ld_b128(lds_dst + t * 16,        src + t * 16);
    async_ld_b128(lds_dst + t * 16 + 4096, src + t * 16 + 4096);
}

// ---------------------------------------------------------------------------
// x: NCHW f32 -> xa: [pixel][channel] bf16, LDS-tiled transpose.
// Block = (one image) x (64 channels): reads one fully contiguous 50KB slab,
// writes 128B-contiguous pixel-major bf16.
// ---------------------------------------------------------------------------
__global__ void __launch_bounds__(256)
convert_x_kernel(const float* __restrict__ x, bf16_t* __restrict__ xa) {
    __shared__ __align__(16) bf16_t tile[HW * 68];   // [hw][c], stride 68 for bank spread
    const int b  = blockIdx.x;        // image
    const int cg = blockIdx.y;        // channel group of 64
    const int t  = threadIdx.x;
    const float* src = x + ((long)b * 1024 + (long)cg * 64) * HW;
    // phase 1: coalesced global reads, scattered LDS writes
    for (int i = t; i < 64 * HW; i += 256) {
        int cq = i / HW, hw = i - cq * HW;
        tile[hw * 68 + cq] = (bf16_t)src[i];
    }
    __syncthreads();
    // phase 2: contiguous 8B stores, 16 threads cover 128B of one pixel row
    bf16_t* dst = xa + (long)(b * HW) * 1024 + cg * 64;
    for (int i = t; i < HW * 16; i += 256) {
        int hw = i >> 4, cv = i & 15;
        v4bf v = *(const v4bf*)&tile[hw * 68 + cv * 4];
        *(v4bf*)(dst + (long)hw * 1024 + cv * 4) = v;
    }
}

// ---------------------------------------------------------------------------
// Pack weights into WMMA B-fragment order (ISA 16-bit B layout, 32x16 KxN):
//   Bp[(((kt*ntiles)+nt)*32 + lane)*16 + j] = B[kt*32 + (lane&16)/2 + (j&7) + (j&8)*2][nt*16 + (lane&15)]
// mode 0: B[k][n] = w[n*K + k]                      (1x1 conv, OIHW)
// mode 1: B[k][n] = w[(n*256 + (k&255))*9 + (k>>8)] (3x3 conv, k = tap*256 + c)
// ---------------------------------------------------------------------------
__global__ void pack_w_kernel(const float* __restrict__ w, bf16_t* __restrict__ bp,
                              int K, int N, int mode) {
    int idx = blockIdx.x * blockDim.x + threadIdx.x;
    if (idx >= K * N) return;
    int j    = idx & 15;
    int lane = (idx >> 4) & 31;
    int rest = idx >> 9;
    int ntiles = N >> 4;
    int nt = rest % ntiles;
    int kt = rest / ntiles;
    int k = kt * 32 + ((lane & 16) >> 1) + (j & 7) + ((j & 8) << 1);
    int n = nt * 16 + (lane & 15);
    float v;
    if (mode == 0) v = w[(long)n * K + k];
    else           v = w[((long)n * 256 + (k & 255)) * 9 + (k >> 8)];
    bp[idx] = (bf16_t)v;
}

// ---------------------------------------------------------------------------
// GEMM1: out1[M,256] = bf16( xa[M,1024] * W1 + (bias1b + bias2a) )
// 8 waves/WG, wave tile 32x32, WG tile 64(M) x 128(N).
// B staged through LDS with double-buffered async copies.
// ---------------------------------------------------------------------------
__global__ void __launch_bounds__(256)
gemm1_kernel(const bf16_t* __restrict__ A, const bf16_t* __restrict__ Bp,
             bf16_t* __restrict__ out, int K, int N,
             const float* __restrict__ s0, const float* __restrict__ s1) {
    __shared__ __align__(32) bf16_t bsm[2][4096];
    const int t = threadIdx.x;
    const int lane = t & 31, wv = t >> 5;
    const int m0 = blockIdx.x * 64  + (wv & 1) * 32;
    const int n0 = blockIdx.y * 128 + (wv >> 1) * 32;
    const int hi = lane >> 4;
    const int ka = hi ? 8 : 0, kb = hi ? 24 : 16;
    const int ntiles = N >> 4;
    const int nt0 = (blockIdx.y * 128) >> 4;          // WG-level first n-tile

    const uint32_t lds0 = (uint32_t)(uint64_t)&bsm[0][0];
    const uint32_t lds1 = (uint32_t)(uint64_t)&bsm[1][0];
    const char* bsrc = (const char*)(Bp + (long)nt0 * 512);
    const long kbytes = (long)ntiles * 1024;          // bytes per k-step, full N

    const bf16_t* ar0 = A + (long)(m0 + (lane & 15)) * K;
    const bf16_t* ar1 = ar0 + (long)16 * K;
    const int g = wv >> 1;
    const int fo0 = (g * 64 + lane) * 16;             // elems into 8KB slice

    v8f acc00 = {}, acc01 = {}, acc10 = {}, acc11 = {};
    stage_btile(lds0, bsrc, t);                       // k-step 0 in flight
    const int nsteps = K >> 5;
    for (int ki = 0; ki < nsteps; ++ki) {
        __syncthreads();                              // prev readers of ping buffer done
        if (ki + 1 < nsteps) {
            stage_btile((ki & 1) ? lds0 : lds1, bsrc + (long)(ki + 1) * kbytes, t);
            asm volatile("s_wait_asynccnt 0x2" ::: "memory");   // step ki complete
        } else {
            asm volatile("s_wait_asynccnt 0x0" ::: "memory");
        }
        __syncthreads();                              // all waves' data visible
        const int kt = ki * 32;
        __builtin_prefetch(ar0 + kt + 64, 0, 3);
        __builtin_prefetch(ar1 + kt + 64, 0, 3);
        v16bf a0 = load_a_frag(ar0 + kt, ka, kb);
        v16bf a1 = load_a_frag(ar1 + kt, ka, kb);
        const bf16_t* bl = &bsm[ki & 1][fo0];
        v16bf b0 = *(const v16bf*)bl;
        v16bf b1 = *(const v16bf*)(bl + 512);
        acc00 = wmma_bf16(a0, b0, acc00);
        acc01 = wmma_bf16(a0, b1, acc01);
        acc10 = wmma_bf16(a1, b0, acc10);
        acc11 = wmma_bf16(a1, b1, acc11);
    }

    const float bias = s0[0] + s1[0];
    const int ncol = lane & 15, mb = hi * 8;
    #pragma unroll
    for (int v = 0; v < 8; ++v) {
        out[(long)(m0 + mb + v)      * N + n0 + ncol]      = (bf16_t)(acc00[v] + bias);
        out[(long)(m0 + mb + v)      * N + n0 + 16 + ncol] = (bf16_t)(acc01[v] + bias);
        out[(long)(m0 + 16 + mb + v) * N + n0 + ncol]      = (bf16_t)(acc10[v] + bias);
        out[(long)(m0 + 16 + mb + v) * N + n0 + 16 + ncol] = (bf16_t)(acc11[v] + bias);
    }
}

// ---------------------------------------------------------------------------
// conv2: implicit 3x3 GEMM, K flattened to 72 uniform 32-wide k-steps
// (tap = ki>>3, each step lies in one tap). OOB neighbors -> zero A fragment.
// Same async double-buffered B staging.
// ---------------------------------------------------------------------------
__global__ void __launch_bounds__(256)
conv2_kernel(const bf16_t* __restrict__ A, const bf16_t* __restrict__ Bp,
             bf16_t* __restrict__ out,
             const float* __restrict__ s0, const float* __restrict__ s1) {
    __shared__ __align__(32) bf16_t bsm[2][4096];
    const int N = 256, Cin = 256;
    const int t = threadIdx.x;
    const int lane = t & 31, wv = t >> 5;
    const int m0 = blockIdx.x * 64  + (wv & 1) * 32;
    const int n0 = blockIdx.y * 128 + (wv >> 1) * 32;
    const int hi = lane >> 4;
    const int ka = hi ? 8 : 0, kb = hi ? 24 : 16;
    const int ntiles = N >> 4;
    const int nt0 = (blockIdx.y * 128) >> 4;

    const uint32_t lds0 = (uint32_t)(uint64_t)&bsm[0][0];
    const uint32_t lds1 = (uint32_t)(uint64_t)&bsm[1][0];
    const char* bsrc = (const char*)(Bp + (long)nt0 * 512);
    const long kbytes = (long)ntiles * 1024;
    const int g = wv >> 1;
    const int fo0 = (g * 64 + lane) * 16;

    const int p0 = m0 + (lane & 15);
    const int p1 = p0 + 16;
    const int img0 = p0 / HW, hw0 = p0 % HW, h0 = hw0 / WD, w0 = hw0 % WD;
    const int img1 = p1 / HW, hw1 = p1 % HW, h1 = hw1 / WD, w1 = hw1 % WD;

    v8f acc00 = {}, acc01 = {}, acc10 = {}, acc11 = {};
    stage_btile(lds0, bsrc, t);
    for (int ki = 0; ki < 72; ++ki) {
        __syncthreads();
        if (ki + 1 < 72) {
            stage_btile((ki & 1) ? lds0 : lds1, bsrc + (long)(ki + 1) * kbytes, t);
            asm volatile("s_wait_asynccnt 0x2" ::: "memory");
        } else {
            asm volatile("s_wait_asynccnt 0x0" ::: "memory");
        }
        __syncthreads();
        const int tap = ki >> 3, kc = (ki & 7) << 5;
        const int ty = tap / 3;
        const int dy = ty - 1, dx = tap - ty * 3 - 1;
        const int hn0 = h0 + dy, wn0 = w0 + dx;
        const int hn1 = h1 + dy, wn1 = w1 + dx;
        const bool ok0 = ((unsigned)hn0 < WD) && ((unsigned)wn0 < WD);
        const bool ok1 = ((unsigned)hn1 < WD) && ((unsigned)wn1 < WD);
        v16bf a0 = {}, a1 = {};
        if (ok0) a0 = load_a_frag(A + (long)(img0 * HW + hn0 * WD + wn0) * Cin + kc, ka, kb);
        if (ok1) a1 = load_a_frag(A + (long)(img1 * HW + hn1 * WD + wn1) * Cin + kc, ka, kb);
        const bf16_t* bl = &bsm[ki & 1][fo0];
        v16bf b0 = *(const v16bf*)bl;
        v16bf b1 = *(const v16bf*)(bl + 512);
        acc00 = wmma_bf16(a0, b0, acc00);
        acc01 = wmma_bf16(a0, b1, acc01);
        acc10 = wmma_bf16(a1, b0, acc10);
        acc11 = wmma_bf16(a1, b1, acc11);
    }

    const float bias = s0[0] + s1[0];
    const int ncol = lane & 15, mb = hi * 8;
    #pragma unroll
    for (int v = 0; v < 8; ++v) {
        out[(long)(m0 + mb + v)      * N + n0 + ncol]      = (bf16_t)(acc00[v] + bias);
        out[(long)(m0 + mb + v)      * N + n0 + 16 + ncol] = (bf16_t)(acc01[v] + bias);
        out[(long)(m0 + 16 + mb + v) * N + n0 + ncol]      = (bf16_t)(acc10[v] + bias);
        out[(long)(m0 + 16 + mb + v) * N + n0 + 16 + ncol] = (bf16_t)(acc11[v] + bias);
    }
}

// ---------------------------------------------------------------------------
// GEMM3 + fused epilogue: out(NCHW f32) = relu((o2*W3 + bias3b)*rscale + rbias + x)
// K = 256, N = 1024. Same async B staging.
// ---------------------------------------------------------------------------
__global__ void __launch_bounds__(256)
gemm3_kernel(const bf16_t* __restrict__ A, const bf16_t* __restrict__ Bp,
             const float* __restrict__ xres, float* __restrict__ out,
             const float* __restrict__ b3b, const float* __restrict__ rscale,
             const float* __restrict__ rbias) {
    __shared__ __align__(32) bf16_t bsm[2][4096];
    const int K = 256, N = 1024;
    const int t = threadIdx.x;
    const int lane = t & 31, wv = t >> 5;
    const int m0 = blockIdx.x * 64  + (wv & 1) * 32;
    const int n0 = blockIdx.y * 128 + (wv >> 1) * 32;
    const int hi = lane >> 4;
    const int ka = hi ? 8 : 0, kb = hi ? 24 : 16;
    const int ntiles = N >> 4;
    const int nt0 = (blockIdx.y * 128) >> 4;

    const uint32_t lds0 = (uint32_t)(uint64_t)&bsm[0][0];
    const uint32_t lds1 = (uint32_t)(uint64_t)&bsm[1][0];
    const char* bsrc = (const char*)(Bp + (long)nt0 * 512);
    const long kbytes = (long)ntiles * 1024;

    const bf16_t* ar0 = A + (long)(m0 + (lane & 15)) * K;
    const bf16_t* ar1 = ar0 + (long)16 * K;
    const int g = wv >> 1;
    const int fo0 = (g * 64 + lane) * 16;

    v8f acc00 = {}, acc01 = {}, acc10 = {}, acc11 = {};
    stage_btile(lds0, bsrc, t);
    const int nsteps = K >> 5;                        // 8
    for (int ki = 0; ki < nsteps; ++ki) {
        __syncthreads();
        if (ki + 1 < nsteps) {
            stage_btile((ki & 1) ? lds0 : lds1, bsrc + (long)(ki + 1) * kbytes, t);
            asm volatile("s_wait_asynccnt 0x2" ::: "memory");
        } else {
            asm volatile("s_wait_asynccnt 0x0" ::: "memory");
        }
        __syncthreads();
        const int kt = ki * 32;
        v16bf a0 = load_a_frag(ar0 + kt, ka, kb);
        v16bf a1 = load_a_frag(ar1 + kt, ka, kb);
        const bf16_t* bl = &bsm[ki & 1][fo0];
        v16bf b0 = *(const v16bf*)bl;
        v16bf b1 = *(const v16bf*)(bl + 512);
        acc00 = wmma_bf16(a0, b0, acc00);
        acc01 = wmma_bf16(a0, b1, acc01);
        acc10 = wmma_bf16(a1, b0, acc10);
        acc11 = wmma_bf16(a1, b1, acc11);
    }

    const float bb_ = b3b[0], rs = rscale[0], rb = rbias[0];
    const int ncol = lane & 15, mb = hi * 8;
    #pragma unroll
    for (int v = 0; v < 8; ++v) {
        #pragma unroll
        for (int tt = 0; tt < 4; ++tt) {
            const int m = m0 + ((tt >> 1) ? 16 : 0) + mb + v;
            const int n = n0 + ((tt & 1) ? 16 : 0) + ncol;
            const float a = (tt == 0) ? acc00[v] : (tt == 1) ? acc01[v]
                          : (tt == 2) ? acc10[v] : acc11[v];
            const int img = m / HW, hw = m % HW;
            const long idx = ((long)img * N + n) * HW + hw;
            const float r = (a + bb_) * rs + rb + xres[idx];
            out[idx] = r > 0.0f ? r : 0.0f;
        }
    }
}

// ---------------------------------------------------------------------------
extern "C" void kernel_launch(void* const* d_in, const int* in_sizes, int n_in,
                              void* d_out, int out_size, void* d_ws, size_t ws_size,
                              hipStream_t stream) {
    const float* x      = (const float*)d_in[0];
    const float* w1     = (const float*)d_in[1];
    const float* w2     = (const float*)d_in[2];
    const float* w3     = (const float*)d_in[3];
    const float* bias1b = (const float*)d_in[4];
    const float* bias2a = (const float*)d_in[5];
    const float* bias2b = (const float*)d_in[6];
    const float* bias3a = (const float*)d_in[7];
    const float* bias3b = (const float*)d_in[8];
    const float* rscale = (const float*)d_in[9];
    const float* rbias  = (const float*)d_in[10];
    float* out = (float*)d_out;

    char* ws = (char*)d_ws;
    auto carve = [&](size_t bytes) {
        char* p = ws;
        ws += (bytes + 255) & ~(size_t)255;
        return p;
    };
    bf16_t* xa  = (bf16_t*)carve((size_t)NPIX * 1024 * 2);
    bf16_t* o1  = (bf16_t*)carve((size_t)NPIX * 256 * 2);
    bf16_t* o2  = (bf16_t*)carve((size_t)NPIX * 256 * 2);
    bf16_t* w1p = (bf16_t*)carve((size_t)1024 * 256 * 2);
    bf16_t* w2p = (bf16_t*)carve((size_t)2304 * 256 * 2);
    bf16_t* w3p = (bf16_t*)carve((size_t)256 * 1024 * 2);

    convert_x_kernel<<<dim3(128, 16), 256, 0, stream>>>(x, xa);
    pack_w_kernel<<<(1024 * 256 + 255) / 256, 256, 0, stream>>>(w1, w1p, 1024, 256, 0);
    pack_w_kernel<<<(2304 * 256 + 255) / 256, 256, 0, stream>>>(w2, w2p, 2304, 256, 1);
    pack_w_kernel<<<(256 * 1024 + 255) / 256, 256, 0, stream>>>(w3, w3p, 256, 1024, 0);

    dim3 blk(256);
    gemm1_kernel<<<dim3(NPIX / 64, 256 / 128), blk, 0, stream>>>(
        xa, w1p, o1, 1024, 256, bias1b, bias2a);
    conv2_kernel<<<dim3(NPIX / 64, 256 / 128), blk, 0, stream>>>(
        o1, w2p, o2, bias2b, bias3a);
    gemm3_kernel<<<dim3(NPIX / 64, 1024 / 128), blk, 0, stream>>>(
        o2, w3p, x, out, bias3b, rscale, rbias);
}
